// SparseMoE_75866302316801
// MI455X (gfx1250) — compile-verified, matrix operand order checked by reading, not run on hardware
//
#include <hip/hip_runtime.h>
#include <hip/hip_bf16.h>
#include <stdint.h>

// Problem constants (from reference)
#define DM   512          // d_model
#define HD   1024         // hidden = 2*DM
#define NE   16           // experts
#define NTOK 8192         // B*S = 4*2048
#define CAP  8192         // per-expert token-list capacity (worst case)

typedef __attribute__((ext_vector_type(16))) __bf16 v16bf;
typedef __attribute__((ext_vector_type(8)))  float  v8f;
typedef __attribute__((ext_vector_type(4)))  int    v4i;

union FragBF { v16bf v; uint4 q[2]; };
static_assert(sizeof(v16bf) == 32, "v16bf must be 32B");

#if __has_builtin(__builtin_amdgcn_global_load_async_to_lds_b128) && \
    __has_builtin(__builtin_amdgcn_s_wait_asynccnt)
#define USE_ASYNC_LDS 1
typedef __attribute__((address_space(1))) v4i gv4i;   // global int4
typedef __attribute__((address_space(3))) v4i lv4i;   // LDS int4
#else
#define USE_ASYNC_LDS 0
#endif

// ---------- helpers ----------
__device__ __forceinline__ unsigned short f2bf(float f) {
  unsigned int u = __float_as_uint(f);
  u += 0x7FFFu + ((u >> 16) & 1u);          // round-to-nearest-even
  return (unsigned short)(u >> 16);
}
__device__ __forceinline__ float bf2f(unsigned short h) {
  return __uint_as_float(((unsigned int)h) << 16);
}
__device__ __forceinline__ unsigned int hashu(unsigned int x) {
  x ^= x >> 16; x *= 0x7feb352du;
  x ^= x >> 15; x *= 0x846ca68bu;
  x ^= x >> 16; return x;
}
__device__ __forceinline__ float noise_norm(unsigned int s) {
  unsigned int a = hashu(s * 0x9E3779B9u + 0x85ebca6bu);
  unsigned int b = hashu(a ^ 0xc2b2ae35u);
  float u1 = (float)(a >> 8) * (1.0f / 16777216.0f) + 1e-7f;
  float u2 = (float)(b >> 8) * (1.0f / 16777216.0f);
  return __fsqrt_rn(-2.0f * __logf(u1)) * __cosf(6.2831853f * u2);
}

// ---------- 1) weight f32 -> bf16 in WMMA fragment layout ----------
// Fragment (32x16 KxN): lane l owns n = l&15; element i (0..15):
//   k = (i<8 ? 0 : 16) + 8*(l>>4) + (i&7)
// w1s frag index f = ((e*64 + nt)*16 + kb), elem addr = f*512 + l*16 + i
// w2s frag index f = ((e*32 + nt)*32 + kb)
__global__ __launch_bounds__(256) void wconv_kernel(
    const float* __restrict__ w1, const float* __restrict__ w2,
    unsigned short* __restrict__ w1s, unsigned short* __restrict__ w2s) {
  unsigned int idx = blockIdx.x * 256u + threadIdx.x;
  if (idx < (1u << 23)) {                       // w1: E*D*H = 2^23 elems
    unsigned int i = idx & 15u, l = (idx >> 4) & 31u;
    unsigned int kb = (idx >> 9) & 15u, nt = (idx >> 13) & 63u, e = idx >> 19;
    unsigned int k = kb * 32u + ((i & 8u) ? 16u : 0u) + 8u * (l >> 4) + (i & 7u);
    unsigned int n = nt * 16u + (l & 15u);
    w1s[idx] = f2bf(w1[(e * 512u + k) * 1024u + n]);
  } else {                                      // w2: E*H*D = 2^23 elems
    unsigned int j = idx - (1u << 23);
    unsigned int i = j & 15u, l = (j >> 4) & 31u;
    unsigned int kb = (j >> 9) & 31u, nt = (j >> 14) & 31u, e = j >> 19;
    unsigned int k = kb * 32u + ((i & 8u) ? 16u : 0u) + 8u * (l >> 4) + (i & 7u);
    unsigned int n = nt * 16u + (l & 15u);
    w2s[j] = f2bf(w2[(e * 1024u + k) * 512u + n]);
  }
}

// ---------- 2) spiking threshold + renormalize -> bf16 ----------
__global__ __launch_bounds__(256) void spike_kernel(
    const float* __restrict__ x, unsigned short* __restrict__ xnb) {
  int wave = threadIdx.x >> 5, lane = threadIdx.x & 31;
  int t = blockIdx.x * 8 + wave;
  const float* xr = x + (size_t)t * DM;
  float vals[16], s = 0.0f;
#pragma unroll
  for (int i = 0; i < 16; ++i) { vals[i] = xr[i * 32 + lane]; s += vals[i]; }
#pragma unroll
  for (int m = 16; m > 0; m >>= 1) s += __shfl_xor(s, m, 32);
  float mean = s * (1.0f / 512.0f);
  // row kept iff mean > 0.1; normalized by (row_sum + eps); else zeros
  float scale = (mean > 0.1f) ? 1.0f / (s + 1e-8f) : 0.0f;
  unsigned short* o = xnb + (size_t)t * DM;
#pragma unroll
  for (int i = 0; i < 16; ++i) o[i * 32 + lane] = f2bf(vals[i] * scale);
}

// ---------- 3) gating: logits, noise, top-2, softmax, expert lists ----------
__global__ __launch_bounds__(256) void gate_kernel(
    const unsigned short* __restrict__ xnb,
    const float* __restrict__ gw, const float* __restrict__ gb,
    float* __restrict__ topi, int* __restrict__ ltok, float* __restrict__ lprob,
    int* __restrict__ cnt, float* __restrict__ usage) {
  int t = blockIdx.x * 256 + threadIdx.x;
  const unsigned short* xr = xnb + (size_t)t * DM;
  float acc[NE];
#pragma unroll
  for (int e = 0; e < NE; ++e) acc[e] = 0.0f;
  for (int d = 0; d < DM; ++d) {
    float xv = bf2f(xr[d]);
    const float* g = gw + d * NE;
#pragma unroll
    for (int e = 0; e < NE; ++e) acc[e] += xv * g[e];
  }
  float v0 = -1e30f, v1 = -1e30f; int i0 = 0, i1 = 0;
#pragma unroll
  for (int e = 0; e < NE; ++e) {
    float l = acc[e] + gb[e] + 0.01f * noise_norm((unsigned)(t * NE + e));
    if (l > v0)      { v1 = v0; i1 = i0; v0 = l; i0 = e; }
    else if (l > v1) { v1 = l;  i1 = e; }
  }
  // softmax over {top1, top2}; the NEG_INF-masked rest contribute exp(-1e9)=0
  float e1 = __expf(v1 - v0);
  float inv = 1.0f / (1.0f + e1);
  float g0 = inv, g1 = e1 * inv;
  topi[t * 2 + 0] = (float)i0;
  topi[t * 2 + 1] = (float)i1;
  int s0 = atomicAdd(&cnt[i0], 1);
  ltok[i0 * CAP + s0] = t; lprob[i0 * CAP + s0] = g0;
  int s1 = atomicAdd(&cnt[i1], 1);
  ltok[i1 * CAP + s1] = t; lprob[i1 * CAP + s1] = g1;
  atomicAdd(&usage[i0], g0);
  atomicAdd(&usage[i1], g1);
}

// ---------- 4) aux load-balancing scalar ----------
__global__ void aux_kernel(const float* __restrict__ usage, float* __restrict__ out) {
  if (threadIdx.x == 0) {
    float s = 0.0f;
    for (int e = 0; e < NE; ++e) s += usage[e];
    float imp[NE], mean = 0.0f;
    for (int e = 0; e < NE; ++e) { imp[e] = usage[e] / (s + 1e-10f); mean += imp[e]; }
    mean *= (1.0f / NE);
    float var = 0.0f;
    for (int e = 0; e < NE; ++e) { float d = imp[e] - mean; var += d * d; }
    var *= (1.0f / NE);
    out[0] = __fsqrt_rn(var) / (mean + 1e-10f);
  }
}

// ---------- 5) sparse expert FFN with bf16 WMMA ----------
// grid = (512 tiles, 16 experts), block = 256 (8 waves).
// Per WG: 16-token tile. GEMM1: [16xDM] x w1[e] -> silu -> h[16xHD] (LDS).
//          GEMM2: h x w2[e] -> [16xDM], scaled by gate prob, atomicAdd to out.
// kb-outer / nt-inner with independent accumulators: A fragment loaded once
// per k-block and reused for 8 (resp. 4) WMMAs -> 8x less LDS traffic and
// 8 independent WMMA dependency chains for the matrix pipe.
__global__ __launch_bounds__(256) void ffn_kernel(
    const unsigned short* __restrict__ xnb,
    const unsigned short* __restrict__ w1s, const unsigned short* __restrict__ w2s,
    const float* __restrict__ b1, const float* __restrict__ b2,
    const int* __restrict__ ltok, const float* __restrict__ lprob,
    const int* __restrict__ cnt, float* __restrict__ out) {
  __shared__ __align__(16) unsigned short xt[16 * DM];   // 16 KB
  __shared__ __align__(16) unsigned short ht[16 * HD];   // 32 KB
  __shared__ int   ttok[16];
  __shared__ float tprob[16];

  const int e = blockIdx.y;
  const int tile = blockIdx.x;
  const int ce = cnt[e];
  if (tile * 16 >= ce) return;                 // uniform per block

  const int tid = threadIdx.x;
  const int lane = tid & 31, wave = tid >> 5;
  const int half = lane >> 4, nlane = lane & 15;

  if (tid < 16) {
    int slot = tile * 16 + tid;
    bool v = slot < ce;
    ttok[tid]  = v ? ltok[e * CAP + slot]  : 0;
    tprob[tid] = v ? lprob[e * CAP + slot] : 0.0f;
  }
  __syncthreads();

  { // stage x tile rows into LDS (each thread: 64B of one row)
    int row = tid >> 4, cg = (tid & 15) * 32;
    const unsigned short* src = xnb + (size_t)ttok[row] * DM + cg;
    unsigned short* dst = xt + row * DM + cg;
#if USE_ASYNC_LDS
#pragma unroll
    for (int j = 0; j < 4; ++j)
      __builtin_amdgcn_global_load_async_to_lds_b128(
          (gv4i*)(src + j * 8), (lv4i*)(dst + j * 8), 0, 0);
    __builtin_amdgcn_s_wait_asynccnt(0);
#else
    const uint4* s4 = (const uint4*)src;
    uint4* d4 = (uint4*)dst;
#pragma unroll
    for (int j = 0; j < 4; ++j) d4[j] = s4[j];
#endif
  }
  __syncthreads();

  // ---- GEMM1: N = HD (64 n-tiles, 8 per wave), K = DM (16 k-blocks) ----
  const unsigned short* w1e = w1s + (size_t)e * (DM * HD);
  {
    v8f acc[8] = {};
    for (int kb = 0; kb < 16; ++kb) {
      FragBF a;
      int ab = nlane * DM + kb * 32 + 8 * half;      // ISA 16-bit A 16x32 layout
      a.q[0] = *(const uint4*)(xt + ab);
      a.q[1] = *(const uint4*)(xt + ab + 16);
#pragma unroll
      for (int j = 0; j < 8; ++j) {
        int nt = wave + 8 * j;
        const unsigned short* bp = w1e + ((size_t)(nt * 16 + kb) * 32 + lane) * 16;
        __builtin_prefetch(bp + 512, 0, 0);          // next k-block, same n-tile
        FragBF b;
        b.q[0] = ((const uint4*)bp)[0];
        b.q[1] = ((const uint4*)bp)[1];
        acc[j] = __builtin_amdgcn_wmma_f32_16x16x32_bf16(
            false, a.v, false, b.v, (short)0, acc[j], false, false);
      }
    }
#pragma unroll
    for (int j = 0; j < 8; ++j) {
      int nt = wave + 8 * j;
      float b1v = b1[e * HD + nt * 16 + nlane];
#pragma unroll
      for (int r = 0; r < 8; ++r) {
        int m = r + 8 * half;
        float h = acc[j][r] + b1v;
        h = h / (1.0f + __expf(-h));                 // SiLU
        ht[m * HD + nt * 16 + nlane] = f2bf(h);
      }
    }
  }
  __syncthreads();

  // ---- GEMM2: N = DM (32 n-tiles, 4 per wave), K = HD (32 k-blocks) ----
  const unsigned short* w2e = w2s + (size_t)e * (HD * DM);
  {
    v8f acc[4] = {};
    for (int kb = 0; kb < 32; ++kb) {
      FragBF a;
      int ab = nlane * HD + kb * 32 + 8 * half;
      a.q[0] = *(const uint4*)(ht + ab);
      a.q[1] = *(const uint4*)(ht + ab + 16);
#pragma unroll
      for (int j = 0; j < 4; ++j) {
        int nt = wave + 8 * j;
        const unsigned short* bp = w2e + ((size_t)(nt * 32 + kb) * 32 + lane) * 16;
        __builtin_prefetch(bp + 512, 0, 0);
        FragBF b;
        b.q[0] = ((const uint4*)bp)[0];
        b.q[1] = ((const uint4*)bp)[1];
        acc[j] = __builtin_amdgcn_wmma_f32_16x16x32_bf16(
            false, a.v, false, b.v, (short)0, acc[j], false, false);
      }
    }
#pragma unroll
    for (int j = 0; j < 4; ++j) {
      int nt = wave + 8 * j;
      int n = nt * 16 + nlane;
      float b2v = b2[e * DM + n];
#pragma unroll
      for (int r = 0; r < 8; ++r) {
        int m = r + 8 * half;
        float val = (acc[j][r] + b2v) * tprob[m];
        atomicAdd(out + (size_t)ttok[m] * DM + n, val);
      }
    }
  }
}

// ---------- launch ----------
extern "C" void kernel_launch(void* const* d_in, const int* in_sizes, int n_in,
                              void* d_out, int out_size, void* d_ws, size_t ws_size,
                              hipStream_t stream) {
  const float* x  = (const float*)d_in[0];
  const float* gw = (const float*)d_in[1];
  const float* gb = (const float*)d_in[2];
  const float* w1 = (const float*)d_in[3];
  const float* b1 = (const float*)d_in[4];
  const float* w2 = (const float*)d_in[5];
  const float* b2 = (const float*)d_in[6];
  float* out = (float*)d_out;

  // workspace layout (bytes)
  const size_t OFF_XNB = 0;                                 // 8,388,608
  const size_t OFF_W1S = OFF_XNB + (size_t)NTOK * DM * 2;   // 16,777,216
  const size_t OFF_W2S = OFF_W1S + (size_t)NE * DM * HD * 2;// 16,777,216
  const size_t OFF_LT  = OFF_W2S + (size_t)NE * HD * DM * 2;// 524,288
  const size_t OFF_LP  = OFF_LT + (size_t)NE * CAP * 4;     // 524,288
  const size_t OFF_CNT = OFF_LP + (size_t)NE * CAP * 4;     // 64
  const size_t OFF_USE = OFF_CNT + 64;                      // 64

  char* ws = (char*)d_ws;
  unsigned short* xnb = (unsigned short*)(ws + OFF_XNB);
  unsigned short* w1s = (unsigned short*)(ws + OFF_W1S);
  unsigned short* w2s = (unsigned short*)(ws + OFF_W2S);
  int*   ltok  = (int*)(ws + OFF_LT);
  float* lprob = (float*)(ws + OFF_LP);
  int*   cnt   = (int*)(ws + OFF_CNT);
  float* usage = (float*)(ws + OFF_USE);

  const size_t OFFI = (size_t)NTOK * DM;    // top_i region in d_out
  (void)hipMemsetAsync(out, 0, OFFI * sizeof(float), stream);  // final accum buffer
  (void)hipMemsetAsync(ws + OFF_CNT, 0, 128, stream);          // cnt + usage

  wconv_kernel<<<65536, 256, 0, stream>>>(w1, w2, w1s, w2s);
  spike_kernel<<<NTOK / 8, 256, 0, stream>>>(x, xnb);
  gate_kernel<<<NTOK / 256, 256, 0, stream>>>(xnb, gw, gb, out + OFFI,
                                              ltok, lprob, cnt, usage);
  aux_kernel<<<1, 32, 0, stream>>>(usage, out + OFFI + (size_t)NTOK * 2);
  ffn_kernel<<<dim3(NTOK / 16, NE), 256, 0, stream>>>(xnb, w1s, w2s, b1, b2,
                                                      ltok, lprob, cnt, out);
}
